// AdaptiveAngleConv_89163521065374
// MI455X (gfx1250) — matrix operand order repaired and checked by reading, not compiled, not run on hardware
//
#include <hip/hip_runtime.h>
#include <hip/hip_bf16.h>

// ---------------------------------------------------------------------------
// AdaptiveAngleConv on MI455X (gfx1250):
//   per angle: deformable bilinear sample (bf16) -> implicit-GEMM 3x3 conv
//   via v_wmma_f32_16x16x32_bf16. B-tiles stream into LDS through the CDNA5
//   async-to-LDS path (global_load_async_to_lds_b128 / ASYNCcnt), A-fragments
//   are pre-packed in per-lane WMMA layout and software-pipelined.
// GEMM: M = 256 out-channels, N = 2*190*190 = 72200 pixels, K = 256*9 = 2304.
// ---------------------------------------------------------------------------

typedef __attribute__((ext_vector_type(16))) __bf16       v16bf;
typedef __attribute__((ext_vector_type(8)))  float        v8f;
typedef __attribute__((ext_vector_type(4)))  unsigned int v4u;

union Frag { v16bf v; v4u u[2]; };

constexpr int kB    = 2;
constexpr int kC    = 256;
constexpr int kH    = 64;
constexpr int kW    = 64;
constexpr int kKS   = 3;
constexpr int kPH   = kH + 2;    // 66 (padded)
constexpr int kPW   = kW + 2;    // 66
constexpr int kEX   = kH * kKS;  // 192 expanded grid
constexpr int kOH   = 190;
constexpr int kOW   = 190;
constexpr int kNPIX = kB * kOH * kOW;        // 72200
constexpr int kKTOT = 9 * kC;                // 2304
constexpr int kNSTEP = kKTOT / 32;           // 72 K-steps
constexpr int kNT   = (kNPIX + 127) / 128;   // 565 N tiles of 128

constexpr float S2  = 1.41421356237309504880f;
constexpr float S2H = 0.70710678118654752440f;

__constant__ float c_ox[5][9] = {
  {0.f,0.f,0.f,0.f,0.f,0.f,0.f,0.f,0.f},
  {1.f-S2, 1.f-S2H, 1.f, -S2H, 0.f, S2H, -1.f, S2H-1.f, S2-1.f},
  {0.f, 1.f, 2.f, -1.f, 0.f, 1.f, -2.f, -1.f, 0.f},
  {1.f, 1.f+S2H, 1.f+S2, -S2H, 0.f, S2H, -1.f-S2, -1.f-S2H, -1.f},
  {2.f, 2.f, 2.f, 0.f, 0.f, 0.f, -2.f, -2.f, -2.f}
};
__constant__ float c_oy[5][9] = {
  {0.f,0.f,0.f,0.f,0.f,0.f,0.f,0.f,0.f},
  {1.f, S2H, S2-1.f, 1.f-S2H, 0.f, S2H-1.f, 1.f-S2, -S2H, -1.f},
  {2.f, 1.f, 0.f, 1.f, 0.f, -1.f, 0.f, -1.f, -2.f},
  {1.f+S2, S2H, -1.f, 1.f+S2H, 0.f, -1.f-S2H, 1.f, -S2H, 1.f+S2},
  {2.f, 0.f, -2.f, 2.f, 0.f, -2.f, 2.f, 0.f, -2.f}
};

__device__ __forceinline__ unsigned short f2bf(float f) {
  unsigned int u = __float_as_uint(f);
  u += 0x7FFFu + ((u >> 16) & 1u);   // round-to-nearest-even
  return (unsigned short)(u >> 16);
}

// ---------------------------------------------------------------------------
// Kernel 1: weight -> bf16, pre-packed A fragments.
// wTp[((mt*72 + step)*32 + lane)*16 + e] = A[M = mt*16 + lane%16]
//                                           [K = step*32 + kofs(lane,e)]
// kofs: lanes 0-15 hold K {0..7, 16..23}; lanes 16-31 hold K {8..15, 24..31}
// (ISA 7.12.2 16-bit A-matrix layout), K = (ky*3+kx)*256 + ci.
// ---------------------------------------------------------------------------
__global__ void aac_wtrans(const float* __restrict__ w, unsigned short* __restrict__ wTp) {
  int idx  = blockIdx.x * 256 + threadIdx.x;   // 0 .. 589823
  int e    = idx & 15;
  int lane = (idx >> 4) & 31;
  int rem  = idx >> 9;
  int step = rem % kNSTEP;
  int mt   = rem / kNSTEP;                     // 0..15
  int K  = step * 32 + ((lane >= 16) ? 8 : 0) + ((e < 8) ? e : e + 8);
  int co = mt * 16 + (lane & 15);
  int g  = K >> 8;
  int ci = K & 255;
  int ky = g / 3, kx = g - 3 * ky;
  wTp[idx] = f2bf(w[(((size_t)co * kC + ci) * kKS + ky) * kKS + kx]);
}

// ---------------------------------------------------------------------------
// Kernel 2: deformable bilinear sample -> xo[b][I][J][ci] (bf16, ci-innermost)
// ---------------------------------------------------------------------------
__global__ void aac_sample(const float* __restrict__ x,
                           unsigned short* __restrict__ xo, int angle) {
  int blk = blockIdx.x;                 // (b*192 + I)*192 + J
  int J = blk % kEX; int t = blk / kEX;
  int I = t % kEX;   int b = t / kEX;
  int ci = threadIdx.x;

  int i = I / kKS, nr = I % kKS;
  int j = J / kKS, nc = J % kKS;
  int n = nr * kKS + nc;

  float pnx = (float)(n / 3 - 1);
  float pny = (float)(n % 3 - 1);
  float px = (float)(i + 1) + pnx + c_ox[angle][n];
  float py = (float)(j + 1) + pny + c_oy[angle][n];

  float fx = floorf(px), fy = floorf(py);
  float qlx = fminf(fmaxf(fx,        0.f), (float)(kPH - 1));
  float qrx = fminf(fmaxf(fx + 1.f,  0.f), (float)(kPH - 1));
  float qly = fminf(fmaxf(fy,        0.f), (float)(kPW - 1));
  float qry = fminf(fmaxf(fy + 1.f,  0.f), (float)(kPW - 1));
  float pxc = fminf(fmaxf(px,        0.f), (float)(kPH - 1));
  float pyc = fminf(fmaxf(py,        0.f), (float)(kPW - 1));

  float glt = (1.f + (qlx - pxc)) * (1.f + (qly - pyc));
  float grb = (1.f - (qrx - pxc)) * (1.f - (qry - pyc));
  float glb = (1.f + (qlx - pxc)) * (1.f - (qry - pyc));
  float grt = (1.f - (qrx - pxc)) * (1.f + (qly - pyc));

  const float* xb = x + ((size_t)b * kC + ci) * (kH * kW);
  auto tap = [&](float qx, float qy) -> float {
    int ix = (int)qx - 1, iy = (int)qy - 1;   // padded -> original coords
    if (ix < 0 || ix >= kH || iy < 0 || iy >= kW) return 0.f;
    return xb[ix * kW + iy];
  };

  float v = glt * tap(qlx, qly) + grb * tap(qrx, qry)
          + glb * tap(qlx, qry) + grt * tap(qrx, qly);
  xo[(size_t)blk * kC + ci] = f2bf(v);
}

// ---------------------------------------------------------------------------
// Kernel 3: implicit-GEMM conv via WMMA bf16.
// Block: 256 threads = 8 wave32. Block tile: M=128 (co) x N=128 (pixels).
// Wave (wm 0..3, wn 0..1): 32x64 -> acc[2][4], 8 wmma / K-step, each B
// fragment reused across 2 A fragments (wmma:ds_load = 1:1).
// B tiles double-buffered in LDS via async-to-LDS; A fragments loaded
// directly from pre-packed global weights, software-pipelined one step ahead.
// ---------------------------------------------------------------------------
__global__ __launch_bounds__(256)
void aac_gemm(const unsigned short* __restrict__ xo,
              const unsigned short* __restrict__ wTp,
              float* __restrict__ out, int angle) {
  __shared__ __align__(16) unsigned short Bls[2][128][32];  // 16 KB, [n][k]

  const int tid  = threadIdx.x;
  const int lane = tid & 31;
  const int wave = tid >> 5;
  const int wm   = wave & 3;               // M subtile pair in block
  const int wn   = wave >> 2;              // N half
  const int n_base = blockIdx.x * 128;
  const int mtg0   = blockIdx.y * 8 + wm * 2;  // global M tiles mtg0, mtg0+1

  // B async-load coords: thread -> (n = tid/2 in tile, 16 contiguous ci)
  const int nb = tid >> 1;                 // 0..127
  const int kq = (tid & 1) * 16;           // 0 or 16
  int nf = n_base + nb; if (nf >= kNPIX) nf = kNPIX - 1;
  const int jj = nf % kOW; int t2 = nf / kOW;
  const int ii = t2 % kOH; const int bb = t2 / kOH;

  v8f acc[2][4];
  #pragma unroll
  for (int ml = 0; ml < 2; ++ml)
    #pragma unroll
    for (int s4 = 0; s4 < 4; ++s4)
      acc[ml][s4] = v8f{0.f,0.f,0.f,0.f,0.f,0.f,0.f,0.f};

  // Issue one 32B async copy (two b128; INST_OFFSET applies to LDS+global).
  auto issueB = [&](int step, int buf) {
    const int k0 = step * 32;
    const int g  = k0 >> 8;
    const int c0 = k0 & 255;
    const int ky = g / 3, kx = g - 3 * ky;
    const unsigned short* src =
        xo + (((size_t)bb * kEX + (ii + ky)) * kEX + (jj + kx)) * kC + c0 + kq;
    unsigned loff = (unsigned)(uintptr_t)&Bls[buf][nb][kq];  // low 32b = LDS addr
    asm volatile("global_load_async_to_lds_b128 %0, %1, off\n\t"
                 "global_load_async_to_lds_b128 %0, %1, off offset:16"
                 :: "v"(loff), "v"(src)
                 : "memory");
  };

  auto loadA = [&](int step, Frag& f0, Frag& f1) {
    const unsigned short* a0 = wTp + (((size_t)mtg0 * kNSTEP + step) * 32 + lane) * 16;
    const unsigned short* a1 = a0 + (size_t)kNSTEP * 512;  // next M tile
    f0.u[0] = *reinterpret_cast<const v4u*>(a0);
    f0.u[1] = *reinterpret_cast<const v4u*>(a0 + 8);
    f1.u[0] = *reinterpret_cast<const v4u*>(a1);
    f1.u[1] = *reinterpret_cast<const v4u*>(a1 + 8);
  };

  issueB(0, 0);
  Frag fa0, fa1;
  loadA(0, fa0, fa1);

  for (int s = 0; s < kNSTEP; ++s) {
    const int cur = s & 1;
    // My async copies for buffer `cur` are done:
    asm volatile("s_wait_asynccnt 0x0" ::: "memory");
    // Everyone's copies done AND everyone finished reading buffer cur^1:
    __syncthreads();
    // Prefetch next K-step's B tile into the other buffer (overlaps WMMAs).
    if (s + 1 < kNSTEP) issueB(s + 1, cur ^ 1);

    // Prefetch next step's A fragments (consumed next iteration).
    Frag na0, na1;
    if (s + 1 < kNSTEP) loadA(s + 1, na0, na1);

    // B fragments for this wave's 4 N subtiles (one dscnt wait covers all).
    const int boff = (lane >= 16) ? 8 : 0;
    Frag fb[4];
    #pragma unroll
    for (int sub = 0; sub < 4; ++sub) {
      const int nrow = wn * 64 + sub * 16 + (lane & 15);
      fb[sub].u[0] = *reinterpret_cast<const v4u*>(&Bls[cur][nrow][boff]);
      fb[sub].u[1] = *reinterpret_cast<const v4u*>(&Bls[cur][nrow][boff + 16]);
    }

    #pragma unroll
    for (int sub = 0; sub < 4; ++sub) {
      acc[0][sub] = __builtin_amdgcn_wmma_f32_16x16x32_bf16(
          false, fa0.v, false, fb[sub].v, (short)0, acc[0][sub], false, false);
      acc[1][sub] = __builtin_amdgcn_wmma_f32_16x16x32_bf16(
          false, fa1.v, false, fb[sub].v, (short)0, acc[1][sub], false, false);
    }
    fa0 = na0;
    fa1 = na1;
  }

  // ---- write D: VGPR r -> M = r + 8*(lane>=16), N = lane%16
  const size_t outA = (size_t)angle * ((size_t)kB * kC * kOH * kOW);
  #pragma unroll
  for (int ml = 0; ml < 2; ++ml) {
    const int mrow0 = (mtg0 + ml) * 16 + ((lane >= 16) ? 8 : 0);
    #pragma unroll
    for (int sub = 0; sub < 4; ++sub) {
      int nfl = n_base + wn * 64 + sub * 16 + (lane & 15);
      if (nfl < kNPIX) {
        int j = nfl % kOW; int t = nfl / kOW;
        int i = t % kOH;   int b = t / kOH;
        #pragma unroll
        for (int r = 0; r < 8; ++r) {
          out[outA + (((size_t)b * kC + (mrow0 + r)) * kOH + i) * kOW + j] =
              acc[ml][sub][r];
        }
      }
    }
  }
}

// ---------------------------------------------------------------------------
// Host launcher. Workspace layout (needs ~38.9 MB):
//   [0)       xo : kB*192*192*256 bf16 (37.75 MB), reused per angle
//   [xo_end)  wTp: 16*72*32*16 bf16 (1.18 MB), WMMA-fragment-packed weights
// Stream ordering serializes sample(a) -> gemm(a) -> sample(a+1), so one xo
// slot is safe to reuse across angles.
// ---------------------------------------------------------------------------
extern "C" void kernel_launch(void* const* d_in, const int* in_sizes, int n_in,
                              void* d_out, int out_size, void* d_ws, size_t ws_size,
                              hipStream_t stream) {
  const float* x = (const float*)d_in[0];
  const float* w = (const float*)d_in[1];
  float* out = (float*)d_out;

  unsigned short* xo  = (unsigned short*)d_ws;
  unsigned short* wTp = xo + (size_t)kB * kEX * kEX * kC;

  aac_wtrans<<<dim3(kKTOT), dim3(256), 0, stream>>>(w, wTp);
  for (int a = 0; a < 5; ++a) {
    aac_sample<<<dim3(kB * kEX * kEX), dim3(kC), 0, stream>>>(x, xo, a);
    aac_gemm<<<dim3(kNT, 2), dim3(256), 0, stream>>>(xo, wTp, out, a);
  }
}
